// GCNwithAttention_24051816858257
// MI455X (gfx1250) — matrix operand order repaired and checked by reading, not compiled, not run on hardware
//
#include <hip/hip_runtime.h>

// ---------------------------------------------------------------------------
// Types for CDNA5 WMMA (wave32): v_wmma_f32_16x16x32_bf16
// ---------------------------------------------------------------------------
typedef __attribute__((ext_vector_type(16))) __bf16 bf16x16;
typedef __attribute__((ext_vector_type(8)))  float  f32x8;

union Pk8    { __bf16 h[8];  uint4  u; };     // 8 bf16 <-> 16B
union Pk4    { __bf16 h[4];  uint2  u; };     // 4 bf16 <-> 8B
union FragBF { uint4  u[2];  bf16x16 v; };    // 16 bf16 fragment <-> 2x16B
union Sc8    { float4 f[2];  float  s[8]; };

// ---------------------------------------------------------------------------
// CDNA5 async DMA: global -> LDS, 16B per lane, tracked by ASYNCcnt.
// No VGPR staging => no register pressure across the WMMA section.
// ---------------------------------------------------------------------------
__device__ __forceinline__ void async_b128(unsigned lds_off, const void* gaddr) {
    asm volatile("global_load_async_to_lds_b128 %0, %1, off"
                 :: "v"(lds_off), "v"(gaddr) : "memory");
}
__device__ __forceinline__ void wait_async0() {
    asm volatile("s_wait_asynccnt 0x0" ::: "memory");
}

// ---------------------------------------------------------------------------
// Generic WMMA GEMM:  C[M x BN] = A[M x K] @ B   (B given transposed Bt[BN][K])
//  Block tile 64 x BN (BN = full output width -> A streamed exactly once).
//  8 waves (2 row x 4 col), wave = 32 x (BN/4) = 2 x NB tiles of 16x16.
//  K-step 32, double-buffered LDS, 1 barrier/step.
//  B tiles (and A tiles when bf16) move via GLOBAL_LOAD_ASYNC_TO_LDS_B128.
//
//  AMODE 0: A bf16 row-major (async DMA)
//  AMODE 1: A fp32 adjacency, on-the-fly d[i]*a*d[j] scale + cvt (fallback)
//  EPI 0: bf16 transposed out_T[f][node] (ldc=M)   EPI 1: relu bf16 row-major
//  EPI 2: fp32 rowmajor Cout + bf16 Cout2          EPI 3: tanh(acc+bias) fp32
//  SC  1: epilogue multiplies acc by disv[nodeBase+e]
//         (EPI0: folds d[j] into B operand; EPI1/2: applies row scale d[i])
// ---------------------------------------------------------------------------
template<int BN, int AMODE, int EPI, int SC>
__global__ __launch_bounds__(256) void gemm_k(
    const void* __restrict__ Aptr, const float* __restrict__ disv,
    const __bf16* __restrict__ Bt, const int K, const int lda, const int ldb,
    void* __restrict__ Cout, void* __restrict__ Cout2,
    const float* __restrict__ bias, const int ldc)
{
    constexpr int NB = BN / 64;                       // WMMA col tiles per wave
    __shared__ __align__(16) unsigned short As[2][64 * 32];
    __shared__ __align__(16) unsigned short Bs[2][BN * 32];

    const int tid  = threadIdx.x;
    const int lane = tid & 31;
    const int wave = tid >> 5;
    const int wrow = wave >> 2;        // 0..1
    const int wcol = wave & 3;         // 0..3
    const int l15  = lane & 15;
    const int kg   = lane >> 4;        // 0..1 (lane group per ISA fragment map)
    const int m0   = blockIdx.x * 64;
    const int nK   = K >> 5;

    f32x8 acc[2][NB] = {};

    const int ar = tid >> 2;           // A tile row   0..63
    const int ac = (tid & 3) * 8;      // A tile col   0,8,16,24

    float4 af0, af1;                   // AMODE1 staging only

    auto asyncB = [&](int ks, int buf) {
#pragma unroll
        for (int i = 0; i < NB; ++i) {
            const int ch = tid * NB + i;
            const int f  = ch >> 2;
            const int ps = (ch & 3) * 8;
            async_b128((unsigned)(uintptr_t)&Bs[buf][f * 32 + ps],
                       Bt + (size_t)f * ldb + ks * 32 + ps);
        }
    };
    auto asyncA = [&](int ks, int buf) {          // AMODE 0
        async_b128((unsigned)(uintptr_t)&As[buf][ar * 32 + ac],
                   (const __bf16*)Aptr + (size_t)(m0 + ar) * lda + ks * 32 + ac);
    };
    auto gloadA = [&](int ks) {                   // AMODE 1: fp32 + scale
        const int k0 = ks * 32;
        const float* ap = (const float*)Aptr + (size_t)(m0 + ar) * lda + k0 + ac;
        af0 = *(const float4*)ap;
        af1 = *(const float4*)(ap + 4);
        const float  sr = disv[m0 + ar];
        const float4 d0 = *(const float4*)(disv + k0 + ac);
        const float4 d1 = *(const float4*)(disv + k0 + ac + 4);
        af0.x *= sr * d0.x; af0.y *= sr * d0.y; af0.z *= sr * d0.z; af0.w *= sr * d0.w;
        af1.x *= sr * d1.x; af1.y *= sr * d1.y; af1.z *= sr * d1.z; af1.w *= sr * d1.w;
    };
    auto sstoreA = [&](int buf) {                 // AMODE 1
        Pk8 p;
        p.h[0] = (__bf16)af0.x; p.h[1] = (__bf16)af0.y;
        p.h[2] = (__bf16)af0.z; p.h[3] = (__bf16)af0.w;
        p.h[4] = (__bf16)af1.x; p.h[5] = (__bf16)af1.y;
        p.h[6] = (__bf16)af1.z; p.h[7] = (__bf16)af1.w;
        *(uint4*)&As[buf][ar * 32 + ac] = p.u;
    };

    // ---- prologue: stage k-step 0 into LDS[0] ----
    asyncB(0, 0);
    if constexpr (AMODE == 0) { asyncA(0, 0); }
    else                      { gloadA(0); sstoreA(0); }
    wait_async0();
    __syncthreads();

    for (int ks = 0; ks < nK; ++ks) {
        const int  cur  = ks & 1;
        const int  nxt  = cur ^ 1;
        const bool more = (ks + 1) < nK;
        if (more) {
            asyncB(ks + 1, nxt);                           // DMA overlaps WMMAs
            if constexpr (AMODE == 0) asyncA(ks + 1, nxt);
            else                      gloadA(ks + 1);
        }

        // Fragment loads from LDS per ISA 16-bit A/B layouts (ds_load_b128).
        FragBF a[2], b[NB];
#pragma unroll
        for (int tr = 0; tr < 2; ++tr) {
            const int r = wrow * 32 + tr * 16 + l15;
            a[tr].u[0] = *(const uint4*)&As[cur][r * 32 + kg * 8];        // K=kg*8..+7
            a[tr].u[1] = *(const uint4*)&As[cur][r * 32 + 16 + kg * 8];   // K=16+kg*8..
        }
#pragma unroll
        for (int tc = 0; tc < NB; ++tc) {
            const int f = wcol * (BN / 4) + tc * 16 + l15;
            const uint4* p = (const uint4*)&Bs[cur][f * 32 + kg * 8];     // K=kg*8..+7, 16+kg*8..
            b[tc].u[0] = p[0];
            b[tc].u[1] = p[2];
        }
#pragma unroll
        for (int tr = 0; tr < 2; ++tr)
#pragma unroll
            for (int tc = 0; tc < NB; ++tc)
                acc[tr][tc] = __builtin_amdgcn_wmma_f32_16x16x32_bf16(
                    false, a[tr].v, false, b[tc].v,
                    (short)0, acc[tr][tc], false, false);

        if (more) {
            if constexpr (AMODE == 1) sstoreA(nxt);
            wait_async0();                 // my DMA landed; barrier covers peers
        }
        __syncthreads();
    }

    // Epilogue. C layout per ISA: col = lane&15, row = e + kg*8 (e = 0..7).
#pragma unroll
    for (int tr = 0; tr < 2; ++tr) {
        const int nodeBase = m0 + wrow * 32 + tr * 16 + kg * 8;
        Sc8 dsc;
        if constexpr (SC) {
            dsc.f[0] = *(const float4*)(disv + nodeBase);
            dsc.f[1] = *(const float4*)(disv + nodeBase + 4);
        }
#pragma unroll
        for (int tc = 0; tc < NB; ++tc) {
            const int f = wcol * (BN / 4) + tc * 16 + l15;
            if constexpr (EPI == 0) {            // bf16, transposed [f][node]
                Pk8 p;
#pragma unroll
                for (int e = 0; e < 8; ++e) {
                    float v = acc[tr][tc][e];
                    if constexpr (SC) v *= dsc.s[e];
                    p.h[e] = (__bf16)v;
                }
                *(uint4*)((__bf16*)Cout + (size_t)f * ldc + nodeBase) = p.u;
            } else if constexpr (EPI == 1) {     // relu -> bf16 row-major
#pragma unroll
                for (int e = 0; e < 8; ++e) {
                    float v = acc[tr][tc][e];
                    if constexpr (SC) v *= dsc.s[e];
                    ((__bf16*)Cout)[(size_t)(nodeBase + e) * ldc + f] =
                        (__bf16)fmaxf(v, 0.0f);
                }
            } else if constexpr (EPI == 2) {     // fp32 out + bf16 copy
#pragma unroll
                for (int e = 0; e < 8; ++e) {
                    float v = acc[tr][tc][e];
                    if constexpr (SC) v *= dsc.s[e];
                    ((float*)Cout)[(size_t)(nodeBase + e) * ldc + f]   = v;
                    ((__bf16*)Cout2)[(size_t)(nodeBase + e) * ldc + f] = (__bf16)v;
                }
            } else {                             // tanh(acc + bias) fp32
                const float bb = bias[f];
#pragma unroll
                for (int e = 0; e < 8; ++e)
                    ((float*)Cout)[(size_t)(nodeBase + e) * ldc + f] =
                        tanhf(acc[tr][tc][e] + bb);
            }
        }
    }
}

// ---------------------------------------------------------------------------
// d[i] = rsqrt(sum_j adj[i][j])      (one block per row, vectorized)
// ---------------------------------------------------------------------------
__global__ __launch_bounds__(256) void rowsum_rsqrt_k(
    const float* __restrict__ A, float* __restrict__ disv, int n)
{
    __shared__ float red[256];
    const int row = blockIdx.x;
    const float4* a4 = (const float4*)(A + (size_t)row * n);
    float s = 0.f;
    for (int j = threadIdx.x; j < (n >> 2); j += 256) {
        const float4 v = a4[j];
        s += (v.x + v.y) + (v.z + v.w);
    }
    red[threadIdx.x] = s;
    __syncthreads();
    for (int off = 128; off > 0; off >>= 1) {
        if (threadIdx.x < off) red[threadIdx.x] += red[threadIdx.x + off];
        __syncthreads();
    }
    if (threadIdx.x == 0) disv[row] = rsqrtf(red[0]);
}

// ---------------------------------------------------------------------------
// fp32 -> bf16 convert, optional transpose (small operands)
// ---------------------------------------------------------------------------
__global__ __launch_bounds__(256) void cvt_bf16_k(
    const float* __restrict__ in, __bf16* __restrict__ out,
    int R, int C, int transpose)
{
    const int total = R * C;
    for (int i = blockIdx.x * 256 + threadIdx.x; i < total; i += gridDim.x * 256) {
        const float v = in[i];
        if (transpose) {
            const int r = i / C, c = i - r * C;
            out[(size_t)c * R + r] = (__bf16)v;
        } else {
            out[i] = (__bf16)v;
        }
    }
}

// Vectorized fp32 -> bf16 for the 256 MB adjacency (float4 in, 4xbf16 out).
__global__ __launch_bounds__(256) void cvt_bf16_vec_k(
    const float* __restrict__ in, __bf16* __restrict__ out, size_t total4)
{
    for (size_t i = (size_t)blockIdx.x * 256 + threadIdx.x; i < total4;
         i += (size_t)gridDim.x * 256) {
        const float4 v = ((const float4*)in)[i];
        Pk4 p;
        p.h[0] = (__bf16)v.x; p.h[1] = (__bf16)v.y;
        p.h[2] = (__bf16)v.z; p.h[3] = (__bf16)v.w;
        ((uint2*)out)[i] = p.u;
    }
}

// ---------------------------------------------------------------------------
// scores[n] = dot(T[n][:], q) + b     (one wave32 per node)
// ---------------------------------------------------------------------------
__global__ __launch_bounds__(256) void score_k(
    const float* __restrict__ T, const float* __restrict__ q,
    const float* __restrict__ bscal, float* __restrict__ sc, int n, int D)
{
    const int w    = (blockIdx.x * 256 + threadIdx.x) >> 5;
    const int lane = threadIdx.x & 31;
    if (w >= n) return;
    const float* t = T + (size_t)w * D;
    float acc = 0.f;
    for (int j = lane; j < D; j += 32) acc += t[j] * q[j];
#pragma unroll
    for (int off = 16; off > 0; off >>= 1) acc += __shfl_down(acc, off, 32);
    if (lane == 0) sc[w] = acc + bscal[0];
}

// ---------------------------------------------------------------------------
// Softmax stats over scores: stats[0]=max, stats[1]=1/sum(exp(s-max))
// ---------------------------------------------------------------------------
__global__ __launch_bounds__(1024) void softmax_stats_k(
    const float* __restrict__ sc, float* __restrict__ stats, int n)
{
    __shared__ float red[1024];
    __shared__ float smx;
    const int t = threadIdx.x;
    float m = -3.402823466e38f;
    for (int i = t; i < n; i += 1024) m = fmaxf(m, sc[i]);
    red[t] = m; __syncthreads();
    for (int off = 512; off > 0; off >>= 1) {
        if (t < off) red[t] = fmaxf(red[t], red[t + off]);
        __syncthreads();
    }
    if (t == 0) smx = red[0];
    __syncthreads();
    const float mx = smx;
    float e = 0.f;
    for (int i = t; i < n; i += 1024) e += expf(sc[i] - mx);
    red[t] = e; __syncthreads();
    for (int off = 512; off > 0; off >>= 1) {
        if (t < off) red[t] += red[t + off];
        __syncthreads();
    }
    if (t == 0) { stats[0] = mx; stats[1] = 1.0f / red[0]; }
}

// ---------------------------------------------------------------------------
// Deterministic two-stage weighted column sum: gc[o] = sum_n attn[n]*z[n][o]
// ---------------------------------------------------------------------------
__global__ __launch_bounds__(256) void wsum_partial_k(
    const float* __restrict__ z, const float* __restrict__ sc,
    const float* __restrict__ stats, float* __restrict__ part,
    int n, int D, int chunks)
{
    __shared__ float red[256];
    const int o    = threadIdx.x % D;
    const int half = threadIdx.x / D;
    const float mx = stats[0], inv = stats[1];
    const int per = n / chunks;
    const int n0  = blockIdx.x * per;
    float acc = 0.f;
    for (int nn = n0 + half; nn < n0 + per; nn += 2) {
        const float wgt = expf(sc[nn] - mx) * inv;
        acc += wgt * z[(size_t)nn * D + o];
    }
    red[threadIdx.x] = acc; __syncthreads();
    if (half == 0) part[blockIdx.x * D + o] = red[o] + red[D + o];
}

__global__ void wsum_final_k(const float* __restrict__ part,
                             float* __restrict__ gc, int P, int D)
{
    const int o = blockIdx.x * blockDim.x + threadIdx.x;
    if (o >= D) return;
    float a = 0.f;
    for (int p = 0; p < P; ++p) a += part[(size_t)p * D + o];
    gc[o] = a;
}

// ---------------------------------------------------------------------------
// Host side
// ---------------------------------------------------------------------------
extern "C" void kernel_launch(void* const* d_in, const int* in_sizes, int n_in,
                              void* d_out, int out_size, void* d_ws, size_t ws_size,
                              hipStream_t stream)
{
    constexpr int N = 8192, IN = 128, H = 256, OUT = 128;
    const float* x   = (const float*)d_in[0];
    const float* adj = (const float*)d_in[1];
    const float* W1  = (const float*)d_in[2];
    const float* W2  = (const float*)d_in[3];
    const float* W3  = (const float*)d_in[4];
    const float* Wl  = (const float*)d_in[5];
    const float* bl  = (const float*)d_in[6];
    const float* q   = (const float*)d_in[7];
    const float* b   = (const float*)d_in[8];
    float* out = (float*)d_out;
    (void)in_sizes; (void)n_in; (void)out_size;

    char* w = (char*)d_ws;
    size_t off = 0;
    auto alloc = [&](size_t bytes) -> void* {
        void* p = w + off;
        off = (off + bytes + 255) & ~(size_t)255;
        return p;
    };
    float*  disv  = (float*) alloc((size_t)N * 4);
    __bf16* xbf   = (__bf16*)alloc((size_t)N * IN * 2);
    __bf16* W1T   = (__bf16*)alloc((size_t)H * IN * 2);
    __bf16* W2T   = (__bf16*)alloc((size_t)H * H * 2);
    __bf16* W3T   = (__bf16*)alloc((size_t)OUT * H * 2);
    __bf16* Wlbf  = (__bf16*)alloc((size_t)OUT * OUT * 2);
    __bf16* fbufT = (__bf16*)alloc((size_t)H * N * 2);
    __bf16* hbuf  = (__bf16*)alloc((size_t)N * H * 2);
    __bf16* zbf   = (__bf16*)alloc((size_t)N * OUT * 2);
    float*  Tbuf  = (float*) alloc((size_t)N * OUT * 4);
    float*  sc    = (float*) alloc((size_t)N * 4);
    float*  stats = (float*) alloc(2 * 4);
    float*  part  = (float*) alloc((size_t)32 * OUT * 4);

    // bf16 adjacency cache (128 MB) if workspace allows: adjbf fits in L2,
    // layers 2+3 re-read it from L2 instead of HBM. Deterministic in ws_size.
    const bool pre = ws_size >= off + (size_t)N * N * 2 + 256;
    __bf16* adjbf = pre ? (__bf16*)alloc((size_t)N * N * 2) : nullptr;

    rowsum_rsqrt_k<<<N, 256, 0, stream>>>(adj, disv, N);
    cvt_bf16_k<<<256, 256, 0, stream>>>(x,  xbf, N, IN, 0);
    cvt_bf16_k<<<64, 256, 0, stream>>>(W1, W1T, IN, H, 1);
    cvt_bf16_k<<<64, 256, 0, stream>>>(W2, W2T, H, H, 1);
    cvt_bf16_k<<<64, 256, 0, stream>>>(W3, W3T, H, OUT, 1);
    cvt_bf16_k<<<64, 256, 0, stream>>>(Wl, Wlbf, OUT, OUT, 0);

    if (pre) {
        // nadj@B = diag(d) * (A_bf @ (diag(d)*B)): fold d[j] into small-GEMM
        // output (SC on EPI0), d[i] into big-GEMM epilogue (SC on EPI1/2).
        cvt_bf16_vec_k<<<2048, 256, 0, stream>>>(adj, adjbf, (size_t)N * N / 4);
        gemm_k<256, 0, 0, 1><<<N / 64, 256, 0, stream>>>(xbf, disv, W1T, IN, IN, IN,
                                                         fbufT, nullptr, nullptr, N);
        gemm_k<256, 0, 1, 1><<<N / 64, 256, 0, stream>>>(adjbf, disv, fbufT, N, N, N,
                                                         hbuf, nullptr, nullptr, H);
        gemm_k<256, 0, 0, 1><<<N / 64, 256, 0, stream>>>(hbuf, disv, W2T, H, H, H,
                                                         fbufT, nullptr, nullptr, N);
        gemm_k<256, 0, 1, 1><<<N / 64, 256, 0, stream>>>(adjbf, disv, fbufT, N, N, N,
                                                         hbuf, nullptr, nullptr, H);
        gemm_k<128, 0, 0, 1><<<N / 64, 256, 0, stream>>>(hbuf, disv, W3T, H, H, H,
                                                         fbufT, nullptr, nullptr, N);
        gemm_k<128, 0, 2, 1><<<N / 64, 256, 0, stream>>>(adjbf, disv, fbufT, N, N, N,
                                                         out, zbf, nullptr, OUT);
    } else {
        // fallback: normalize adjacency on the fly from fp32 (AMODE 1)
        gemm_k<256, 0, 0, 0><<<N / 64, 256, 0, stream>>>(xbf, nullptr, W1T, IN, IN, IN,
                                                         fbufT, nullptr, nullptr, N);
        gemm_k<256, 1, 1, 0><<<N / 64, 256, 0, stream>>>(adj, disv, fbufT, N, N, N,
                                                         hbuf, nullptr, nullptr, H);
        gemm_k<256, 0, 0, 0><<<N / 64, 256, 0, stream>>>(hbuf, nullptr, W2T, H, H, H,
                                                         fbufT, nullptr, nullptr, N);
        gemm_k<256, 1, 1, 0><<<N / 64, 256, 0, stream>>>(adj, disv, fbufT, N, N, N,
                                                         hbuf, nullptr, nullptr, H);
        gemm_k<128, 0, 0, 0><<<N / 64, 256, 0, stream>>>(hbuf, nullptr, W3T, H, H, H,
                                                         fbufT, nullptr, nullptr, N);
        gemm_k<128, 1, 2, 0><<<N / 64, 256, 0, stream>>>(adj, disv, fbufT, N, N, N,
                                                         out, zbf, nullptr, OUT);
    }

    // attention pooling (common tail)
    gemm_k<128, 0, 3, 0><<<N / 64, 256, 0, stream>>>(zbf, nullptr, Wlbf, OUT, OUT, OUT,
                                                     Tbuf, nullptr, bl, OUT);
    score_k<<<(N * 32) / 256, 256, 0, stream>>>(Tbuf, q, b, sc, N, OUT);
    softmax_stats_k<<<1, 1024, 0, stream>>>(sc, stats, N);
    wsum_partial_k<<<32, 256, 0, stream>>>(out, sc, stats, part, N, OUT, 32);
    wsum_final_k<<<1, 128, 0, stream>>>(part, out + (size_t)N * OUT, 32, OUT);
}
